// ResidualTransformerBlockDropout_23837068493501
// MI455X (gfx1250) — matrix operand order repaired and checked by reading, not compiled, not run on hardware
//
#include <hip/hip_runtime.h>
#include <cstdint>
#include <cstddef>
#include <cmath>

// ---------------- CDNA5 WMMA types ----------------
typedef __attribute__((ext_vector_type(16))) __bf16 bf16x16;
typedef __attribute__((ext_vector_type(8)))  __bf16 bf16x8;
typedef __attribute__((ext_vector_type(8)))  float  f32x8;

__device__ __forceinline__ __bf16 f2bf(float f) {
  union { float f; uint32_t u; } v; v.f = f;
  uint32_t r = v.u + 0x7FFFu + ((v.u >> 16) & 1u);   // round-to-nearest-even
  uint16_t h = (uint16_t)(r >> 16);
  __bf16 o; __builtin_memcpy(&o, &h, 2); return o;
}

// ---- CDNA5 async global->LDS fill (GVS mode: SGPR64 base + per-lane u32 byte offset) ----
// Tracked with ASYNCcnt (CDNA5 ISA); bypasses VGPR staging.
__device__ __forceinline__ void async_load_b128(void* lds_ptr, const void* base, uint32_t byte_off) {
  const uint32_t lds_addr = (uint32_t)(uintptr_t)lds_ptr;
  asm volatile("global_load_async_to_lds_b128 %0, %1, %2"
               :
               : "v"(lds_addr), "v"(byte_off), "s"(base)
               : "memory");
}
__device__ __forceinline__ void wait_async0() {
  asm volatile("s_wait_asynccnt 0x0" ::: "memory");
}

// A fragment (16x32 bf16): lane<16 -> M=lane, K = 0..7 & 16..23 ; lane>=16 -> K = 8..15 & 24..31
__device__ __forceinline__ bf16x16 load_frag_a(const __bf16* row, int hi) {
  bf16x8 lo  = *(const bf16x8*)(row + hi * 8);
  bf16x8 hi8 = *(const bf16x8*)(row + 16 + hi * 8);
  bf16x16 r;
#pragma unroll
  for (int i = 0; i < 8; ++i) { r[i] = lo[i]; r[i + 8] = hi8[i]; }
  return r;
}

// B fragment (32x16 bf16), staged as B^T[n][k]: lane<16 -> N=lane, K = 0..7 & 16..23 ; lane>=16 -> K = 8..15 & 24..31
__device__ __forceinline__ bf16x16 load_frag_b(const __bf16* row, int hi) {
  bf16x8 lo  = *(const bf16x8*)(row + hi * 8);
  bf16x8 hi8 = *(const bf16x8*)(row + 16 + hi * 8);
  bf16x16 r;
#pragma unroll
  for (int i = 0; i < 8; ++i) { r[i] = lo[i]; r[i + 8] = hi8[i]; }
  return r;
}

__device__ __forceinline__ f32x8 wmma_bf16(bf16x16 a, bf16x16 b, f32x8 c) {
  return __builtin_amdgcn_wmma_f32_16x16x32_bf16(false, a, false, b, (short)0, c, false, false);
}

// ---------------- LayerNorm -> bf16 ----------------
__global__ __launch_bounds__(256) void layernorm_bf16_kernel(
    const float* __restrict__ x, const float* __restrict__ g,
    const float* __restrict__ b, __bf16* __restrict__ out, int C) {
  const int row = blockIdx.x;
  const int tid = threadIdx.x;
  const float4 v = ((const float4*)(x + (size_t)row * C))[tid];   // C == 4*blockDim
  __shared__ float red[8];
  float s = v.x + v.y + v.z + v.w;
#pragma unroll
  for (int o = 16; o > 0; o >>= 1) s += __shfl_xor(s, o, 32);
  if ((tid & 31) == 0) red[tid >> 5] = s;
  __syncthreads();
  float mu = 0.f;
#pragma unroll
  for (int i = 0; i < 8; ++i) mu += red[i];
  mu *= (1.0f / (float)C);
  const float d0 = v.x - mu, d1 = v.y - mu, d2 = v.z - mu, d3 = v.w - mu;
  float ss = d0 * d0 + d1 * d1 + d2 * d2 + d3 * d3;
#pragma unroll
  for (int o = 16; o > 0; o >>= 1) ss += __shfl_xor(ss, o, 32);
  __syncthreads();
  if ((tid & 31) == 0) red[tid >> 5] = ss;
  __syncthreads();
  float var = 0.f;
#pragma unroll
  for (int i = 0; i < 8; ++i) var += red[i];
  var *= (1.0f / (float)C);
  const float rstd = rsqrtf(var + 1e-5f);
  const float4 gv = ((const float4*)g)[tid];
  const float4 bv = ((const float4*)b)[tid];
  __bf16* o4 = out + (size_t)row * C + tid * 4;
  o4[0] = f2bf(d0 * rstd * gv.x + bv.x);
  o4[1] = f2bf(d1 * rstd * gv.y + bv.y);
  o4[2] = f2bf(d2 * rstd * gv.z + bv.z);
  o4[3] = f2bf(d3 * rstd * gv.w + bv.w);
}

// ---------------- Weight converters (fp32 -> bf16, pre-transposed to [N,K]) ----------------
__global__ void transpose_f32_to_bf16(const float* __restrict__ src,  // [K,N]
                                      __bf16* __restrict__ dst,       // [N,K]
                                      int K, int N) {
  size_t i = (size_t)blockIdx.x * 256 + threadIdx.x;
  if (i >= (size_t)K * N) return;
  size_t n = i / K, k = i % K;
  dst[i] = f2bf(src[k * (size_t)N + n]);
}

// wq/wk/wv: [H, C, 64] -> WqkvT [3C, C], row j = output feature (q | k | v stacked)
__global__ void qkv_weights_to_bf16(const float* __restrict__ wq, const float* __restrict__ wk,
                                    const float* __restrict__ wv, __bf16* __restrict__ dst, int C) {
  size_t i = (size_t)blockIdx.x * 256 + threadIdx.x;
  if (i >= (size_t)3 * C * C) return;
  int c = (int)(i % C);
  size_t j = i / C;
  int which = (int)(j / C);
  int hd = (int)(j % C);
  int h = hd >> 6, d = hd & 63;
  const float* w = (which == 0) ? wq : ((which == 1) ? wk : wv);
  dst[i] = f2bf(w[((size_t)h * C + c) * 64 + d]);
}

// ---------------- bf16 GEMM: C = A[M,K] @ BT[N,K]^T (+bias)(+res)(relu), f32 or bf16 out ----
// LDS tiles filled with GLOBAL_LOAD_ASYNC_TO_LDS_B128 (ASYNCcnt), double-buffered so the
// next fill overlaps the current tile's 16 WMMAs.
template <bool BIAS, bool RES, bool RELU, bool OUTBF>
__global__ __launch_bounds__(256) void gemm_bf16_kernel(
    const __bf16* __restrict__ A, const __bf16* __restrict__ BT,
    const float* __restrict__ bias, const float* __restrict__ res,
    float* __restrict__ Cf, __bf16* __restrict__ Cb,
    int M, int N, int K, int ldc) {
  __shared__ __bf16 sA[2][128 * 72];   // 128 rows x 64 K, stride 72 (conflict-free, 16B aligned)
  __shared__ __bf16 sB[2][128 * 72];

  const int tid = threadIdx.x;
  const int wave = tid >> 5, lane = tid & 31;
  const int l = lane & 15, hi = lane >> 4;
  const int m0 = blockIdx.y * 128;
  const int n0 = blockIdx.x * 128;
  const int wm = (wave >> 1) * 32;     // 4x2 wave grid -> 32x64 wave tile
  const int wn = (wave & 1) * 64;

  f32x8 acc[2][4] = {};

  int rrow[4], rcol[4];
#pragma unroll
  for (int i = 0; i < 4; ++i) { int c = tid + i * 256; rrow[i] = c >> 3; rcol[i] = (c & 7) * 8; }

  auto issue_tile = [&](int k0, int buf) {
#pragma unroll
    for (int i = 0; i < 4; ++i) {
      const uint32_t aoff = (uint32_t)((((size_t)(m0 + rrow[i]) * K) + k0 + rcol[i]) * 2);
      const uint32_t boff = (uint32_t)((((size_t)(n0 + rrow[i]) * K) + k0 + rcol[i]) * 2);
      async_load_b128(&sA[buf][rrow[i] * 72 + rcol[i]], A, aoff);
      async_load_b128(&sB[buf][rrow[i] * 72 + rcol[i]], BT, boff);
      if (k0 + 128 <= K) {               // warm L2 for the tile after next (global_prefetch_b8)
        __builtin_prefetch((const char*)A + aoff + 128, 0, 1);
        __builtin_prefetch((const char*)BT + boff + 128, 0, 1);
      }
    }
  };
  auto compute = [&](int buf) {
#pragma unroll
    for (int ks = 0; ks < 64; ks += 32) {
      bf16x16 af[2], bfv[4];
#pragma unroll
      for (int mi = 0; mi < 2; ++mi) af[mi] = load_frag_a(&sA[buf][(wm + mi * 16 + l) * 72 + ks], hi);
#pragma unroll
      for (int ni = 0; ni < 4; ++ni) bfv[ni] = load_frag_b(&sB[buf][(wn + ni * 16 + l) * 72 + ks], hi);
#pragma unroll
      for (int mi = 0; mi < 2; ++mi)
#pragma unroll
        for (int ni = 0; ni < 4; ++ni)
          acc[mi][ni] = wmma_bf16(af[mi], bfv[ni], acc[mi][ni]);
    }
  };

  const int nk = K >> 6;
  issue_tile(0, 0);
  wait_async0();
  __syncthreads();
  int buf = 0;
  for (int kt = 0; kt < nk; ++kt) {
    // fill buf^1 while computing buf: buf^1 was last read in iteration kt-1, and every wave
    // passed that iteration's closing barrier, so the async writes cannot race a reader.
    if (kt + 1 < nk) issue_tile((kt + 1) * 64, buf ^ 1);
    compute(buf);
    if (kt + 1 < nk) {
      wait_async0();
      __syncthreads();
      buf ^= 1;
    }
  }

  // epilogue: C-layout -> lane holds col (n*16+l), rows hi*8+r
#pragma unroll
  for (int mi = 0; mi < 2; ++mi) {
#pragma unroll
    for (int ni = 0; ni < 4; ++ni) {
      const int col = n0 + wn + ni * 16 + l;
      const float bcol = BIAS ? bias[col] : 0.0f;
#pragma unroll
      for (int r = 0; r < 8; ++r) {
        const int row = m0 + wm + mi * 16 + hi * 8 + r;
        float v = acc[mi][ni][r];
        if (BIAS) v += bcol;
        if (RES)  v += res[(size_t)row * ldc + col];
        if (RELU) v = fmaxf(v, 0.0f);
        if (OUTBF) Cb[(size_t)row * ldc + col] = f2bf(v);
        else       Cf[(size_t)row * ldc + col] = v;
      }
    }
  }
}

// ---------------- Flash attention (causal), bf16 WMMA, f32 softmax ----------------
// qkv: [B*T, 3C] bf16 (q | k | v per 1024 cols, head h at h*64). out: [B*T, C] bf16.
__global__ __launch_bounds__(256) void attention_kernel(
    const __bf16* __restrict__ qkv, __bf16* __restrict__ out) {
  constexpr int T = 1024, C = 1024, HS = 64, TB = 128;
  constexpr int LD3 = 3 * C;
  const int tb = blockIdx.x & 7;
  const int h  = (blockIdx.x >> 3) & 15;
  const int b  = blockIdx.x >> 7;

  __shared__ __bf16 sq[128 * 72];        // q rows [t][d]
  __shared__ __bf16 sk[128 * 72];        // k rows [s][d]  == B^T layout for S
  __shared__ __bf16 svT[64 * 136];       // v^T [d][s]     == B^T layout for P@V
  __shared__ __bf16 sp[8 * 16 * 136];    // per-wave P [16][128]

  const int tid = threadIdx.x;
  const int wave = tid >> 5, lane = tid & 31;
  const int l = lane & 15, hi = lane >> 4;
  const size_t rowQ = (size_t)b * T + tb * TB;

  // q tile -> LDS via async fill (completion folded into the first k-block wait)
#pragma unroll
  for (int i = 0; i < 4; ++i) {
    int cidx = tid + i * 256;
    int r = cidx >> 3, c8 = (cidx & 7) * 8;
    async_load_b128(&sq[r * 72 + c8], qkv,
                    (uint32_t)(((rowQ + r) * LD3 + h * HS + c8) * 2));
  }

  float mrow[8], lrow[8];
  f32x8 Oacc[4] = {};
#pragma unroll
  for (int r = 0; r < 8; ++r) { mrow[r] = -INFINITY; lrow[r] = 0.f; }

  __bf16* pw = &sp[wave * 16 * 136];
  const float scale = 0.03125f;          // C^{-1/2} = 1024^{-1/2}

  for (int sb = 0; sb <= tb; ++sb) {
    __syncthreads();                     // previous iteration done reading sk/svT
    const size_t rowK = (size_t)b * T + sb * TB;
#pragma unroll
    for (int i = 0; i < 4; ++i) {        // k tile: async global->LDS
      int cidx = tid + i * 256;
      int r = cidx >> 3, c8 = (cidx & 7) * 8;
      async_load_b128(&sk[r * 72 + c8], qkv,
                      (uint32_t)(((rowK + r) * LD3 + C + h * HS + c8) * 2));
    }
#pragma unroll
    for (int i = 0; i < 32; ++i) {       // v transpose into LDS (element permute -> VALU path)
      int e = tid + i * 256;
      int d = e & 63, s = e >> 6;
      svT[d * 136 + s] = qkv[(rowK + s) * LD3 + 2 * C + h * HS + d];
    }
    wait_async0();                       // q (first iter) + k fills complete
    __syncthreads();

    // S = q_wave(16 rows) @ k^T : 1 M-tile x 8 N-tiles, K=64 in 2 steps
    f32x8 sc[8] = {};
#pragma unroll
    for (int ks = 0; ks < 64; ks += 32) {
      bf16x16 a = load_frag_a(&sq[(wave * 16 + l) * 72 + ks], hi);
#pragma unroll
      for (int n = 0; n < 8; ++n) {
        bf16x16 bb = load_frag_b(&sk[(n * 16 + l) * 72 + ks], hi);
        sc[n] = wmma_bf16(a, bb, sc[n]);
      }
    }

    const bool diag = (sb == tb);
    const int t0 = tb * TB + wave * 16 + hi * 8;
#pragma unroll
    for (int n = 0; n < 8; ++n) {
      const int scol = sb * TB + n * 16 + l;
#pragma unroll
      for (int r = 0; r < 8; ++r) {
        float v = sc[n][r] * scale;
        if (diag && scol > t0 + r) v = -INFINITY;
        sc[n][r] = v;
      }
    }

    // row-wise online softmax (row lives in one half-wave: 4 xor-shuffles)
    float mnew[8], alpha[8];
#pragma unroll
    for (int r = 0; r < 8; ++r) {
      float m = sc[0][r];
#pragma unroll
      for (int n = 1; n < 8; ++n) m = fmaxf(m, sc[n][r]);
      m = fmaxf(m, __shfl_xor(m, 1, 32));
      m = fmaxf(m, __shfl_xor(m, 2, 32));
      m = fmaxf(m, __shfl_xor(m, 4, 32));
      m = fmaxf(m, __shfl_xor(m, 8, 32));
      mnew[r] = fmaxf(mrow[r], m);
      alpha[r] = __expf(mrow[r] - mnew[r]);
      mrow[r] = mnew[r];
    }
#pragma unroll
    for (int n = 0; n < 8; ++n)
#pragma unroll
      for (int r = 0; r < 8; ++r) sc[n][r] = __expf(sc[n][r] - mnew[r]);
#pragma unroll
    for (int r = 0; r < 8; ++r) {
      float s = 0.f;
#pragma unroll
      for (int n = 0; n < 8; ++n) s += sc[n][r];
      s += __shfl_xor(s, 1, 32);
      s += __shfl_xor(s, 2, 32);
      s += __shfl_xor(s, 4, 32);
      s += __shfl_xor(s, 8, 32);
      lrow[r] = lrow[r] * alpha[r] + s;
    }
#pragma unroll
    for (int ni = 0; ni < 4; ++ni)
#pragma unroll
      for (int r = 0; r < 8; ++r) Oacc[ni][r] *= alpha[r];

    // P (C-layout) -> per-wave LDS row-major, then consume as A fragments
#pragma unroll
    for (int n = 0; n < 8; ++n)
#pragma unroll
      for (int r = 0; r < 8; ++r)
        pw[(hi * 8 + r) * 136 + n * 16 + l] = f2bf(sc[n][r]);

    // O += P[16,128] @ V[128,64]
#pragma unroll
    for (int ks = 0; ks < 128; ks += 32) {
      bf16x16 a = load_frag_a(&pw[l * 136 + ks], hi);
#pragma unroll
      for (int ni = 0; ni < 4; ++ni) {
        bf16x16 bb = load_frag_b(&svT[(ni * 16 + l) * 136 + ks], hi);
        Oacc[ni] = wmma_bf16(a, bb, Oacc[ni]);
      }
    }
  }

  // normalize + store concat-head output (bf16, row-major [B*T, C])
#pragma unroll
  for (int r = 0; r < 8; ++r) {
    const float invl = 1.0f / lrow[r];
    const size_t row = rowQ + wave * 16 + hi * 8 + r;
#pragma unroll
    for (int ni = 0; ni < 4; ++ni)
      out[row * C + h * HS + ni * 16 + l] = f2bf(Oacc[ni][r] * invl);
  }
}

// ---------------- launch ----------------
extern "C" void kernel_launch(void* const* d_in, const int* in_sizes, int n_in,
                              void* d_out, int out_size, void* d_ws, size_t ws_size,
                              hipStream_t stream) {
  (void)in_sizes; (void)n_in; (void)out_size; (void)ws_size;
  constexpr int B = 8, T = 1024, C = 1024, H = 16;
  constexpr int M = B * T;
  constexpr int C3 = 3 * C, C4 = 4 * C;

  const float* x      = (const float*)d_in[0];
  const float* wq     = (const float*)d_in[1];
  const float* wk     = (const float*)d_in[2];
  const float* wv     = (const float*)d_in[3];
  const float* w_proj = (const float*)d_in[4];
  const float* b_proj = (const float*)d_in[5];
  const float* ln1_g  = (const float*)d_in[6];
  const float* ln1_b  = (const float*)d_in[7];
  const float* ln2_g  = (const float*)d_in[8];
  const float* ln2_b  = (const float*)d_in[9];
  const float* w1     = (const float*)d_in[10];
  const float* b1     = (const float*)d_in[11];
  const float* w2     = (const float*)d_in[12];
  const float* b2     = (const float*)d_in[13];

  char* ws = (char*)d_ws;
  size_t off = 0;
  auto take = [&](size_t bytes) -> char* {
    char* p = ws + off;
    off += (bytes + 255) & ~(size_t)255;
    return p;
  };
  __bf16* h1b    = (__bf16*)take((size_t)M * C * 2);    // LN1(x)
  __bf16* WqkvT  = (__bf16*)take((size_t)C3 * C * 2);
  __bf16* WprojT = (__bf16*)take((size_t)C * C * 2);
  __bf16* W1T    = (__bf16*)take((size_t)C4 * C * 2);
  __bf16* W2T    = (__bf16*)take((size_t)C * C4 * 2);
  __bf16* qkvb   = (__bf16*)take((size_t)M * C3 * 2);   // fused QKV
  __bf16* aout   = (__bf16*)take((size_t)M * C * 2);    // attention out (concat heads)
  float*  x1     = (float*)take((size_t)M * C * 4);     // x + attn proj
  __bf16* h2b    = (__bf16*)take((size_t)M * C * 2);    // LN2(x1)
  __bf16* rb     = (__bf16*)take((size_t)M * C4 * 2);   // relu(h2@w1+b1)

  {
    int n;
    n = C * C;   transpose_f32_to_bf16<<<(n + 255) / 256, 256, 0, stream>>>(w_proj, WprojT, C, C);
    n = C * C4;  transpose_f32_to_bf16<<<(n + 255) / 256, 256, 0, stream>>>(w1, W1T, C, C4);
    n = C4 * C;  transpose_f32_to_bf16<<<(n + 255) / 256, 256, 0, stream>>>(w2, W2T, C4, C);
    n = C3 * C;  qkv_weights_to_bf16<<<(n + 255) / 256, 256, 0, stream>>>(wq, wk, wv, WqkvT, C);
  }

  layernorm_bf16_kernel<<<M, 256, 0, stream>>>(x, ln1_g, ln1_b, h1b, C);

  gemm_bf16_kernel<false, false, false, true>
      <<<dim3(C3 / 128, M / 128), 256, 0, stream>>>(h1b, WqkvT, nullptr, nullptr,
                                                    nullptr, qkvb, M, C3, C, C3);

  attention_kernel<<<B * H * (T / 128), 256, 0, stream>>>(qkvb, aout);

  gemm_bf16_kernel<true, true, false, false>
      <<<dim3(C / 128, M / 128), 256, 0, stream>>>(aout, WprojT, b_proj, x,
                                                   x1, nullptr, M, C, C, C);

  layernorm_bf16_kernel<<<M, 256, 0, stream>>>(x1, ln2_g, ln2_b, h2b, C);

  gemm_bf16_kernel<true, false, true, true>
      <<<dim3(C4 / 128, M / 128), 256, 0, stream>>>(h2b, W1T, b1, nullptr,
                                                    nullptr, rb, M, C4, C, C4);

  gemm_bf16_kernel<true, true, false, false>
      <<<dim3(C / 128, M / 128), 256, 0, stream>>>(rb, W2T, b2, x1,
                                                   (float*)d_out, nullptr, M, C, C4, C);
}
